// WeatherModel_52003464020214
// MI455X (gfx1250) — matrix operand order repaired and checked by reading, not compiled, not run on hardware
//
#include <hip/hip_runtime.h>
#include <hip/hip_bf16.h>

typedef __attribute__((ext_vector_type(16))) __bf16 v16bf;
typedef __attribute__((ext_vector_type(8)))  float  v8f;
typedef __attribute__((ext_vector_type(4)))  unsigned int u32x4;
typedef __attribute__((ext_vector_type(8)))  int i32x8;
typedef __attribute__((ext_vector_type(4)))  int i32x4;

#define Bn    16
#define TIN   12
#define TOUT  6
#define Dn    10
#define Hn    96
#define Wpx   96
#define HWn   (Hn*Wpx)
#define HIDn  64
#define Mtot  (Bn*HWn)

#define BM 128
#define BN 64
#define BK 64
#define LDA 68   // 64 bf16 data + 4 bf16 pad (8B) per row; 136B = 34 banks
#define LDB 68

__device__ __forceinline__ unsigned short f2bf(float x) {
  unsigned int u = __builtin_bit_cast(unsigned int, x);
  unsigned int r = (u + 0x7FFFu + ((u >> 16) & 1u)) >> 16;
  return (unsigned short)r;
}

__device__ __forceinline__ float sigf(float x) { return 1.f / (1.f + __expf(-x)); }

// ---------------------------------------------------------------------------
// Implicit-GEMM 3x3 SAME conv via v_wmma_f32_16x16x32_bf16.
// A (M=BHW x K=Cin*9): gathered activations (up to 3 concatenated NCHW tensors),
// staged f32->bf16 into padded LDS. B: bf16 weights [N][Ktot], staged by the
// Tensor Data Mover (tensor_load_to_lds) with TDM padding producing the same
// 68-halfword LDS row stride; OOB rows/cols of the tile are zero-filled by TDM.
// ---------------------------------------------------------------------------
__global__ void __launch_bounds__(256)
conv_wmma(const unsigned short* __restrict__ Wb, const float* __restrict__ bias,
          float* __restrict__ out, int out_bs, int N, int Ktot, int act,
          const float* __restrict__ p0, int C0, int cs0, int bs0,
          const float* __restrict__ p1, int C1, int cs1, int bs1,
          const float* __restrict__ p2, int C2, int cs2, int bs2)
{
  __shared__ unsigned short sA[BM * LDA];   // 17408 B
  __shared__ unsigned short sB[BN * LDB];   //  8704 B

  const int tid  = threadIdx.x;
  const int lane = tid & 31;
  const int wave = tid >> 5;
  const int wm   = wave & 3;   // 4 waves over M
  const int wn   = wave >> 2;  // 2 waves over N
  const int m0   = blockIdx.x * BM;
  const int n0   = blockIdx.y * BN;
  const int C01  = C0 + C1;

  // 9216 % 128 == 0 -> a block never crosses a batch image
  const int bimg = m0 / HWn;
  const int pix0 = m0 - bimg * HWn;

  // A-gather thread mapping: one k column per thread, 32 rows (+4 pixel stride)
  const int kk = tid >> 2;   // 0..63
  const int rg = tid & 3;

  v8f acc[2][2];
  for (int i = 0; i < 2; ++i)
    for (int j = 0; j < 2; ++j)
      for (int e = 0; e < 8; ++e) acc[i][j][e] = 0.f;

  const unsigned sB_lds = (unsigned)(unsigned long long)(void*)sB;

  for (int k0 = 0; k0 < Ktot; k0 += BK) {
    // ---- issue TDM load of the B tile (wave 0 only; EXEC-uniform branch) ----
    if (wave == 0) {
      const unsigned long long ga =
          (unsigned long long)(const void*)(Wb + (size_t)n0 * Ktot + k0);
      const int td0 = Ktot - k0;   // remaining K from tile start (OOB -> zero)
      const int td1 = N - n0;      // remaining N rows from tile start
      u32x4 g0;
      g0[0] = 1u;                                   // count=1 valid descriptor
      g0[1] = sB_lds;                               // lds_addr
      g0[2] = (unsigned)ga;                         // global_addr[31:0]
      g0[3] = (unsigned)((ga >> 32) & 0x1FFFFFFu)   // global_addr[56:32]
              | 0x80000000u;                        // type=2 ("image")
      i32x8 g1;
      g1[0] = (1 << 16)      // data_size = 2 bytes
            | (1 << 20)      // pad_enable
            | (4 << 22)      // pad_interval: 32 DWORDs (=64 bf16 data)
            | (1 << 25);     // pad_amount: 2 DWORDs (=4 bf16 pad) -> stride 68
      g1[1] = (td0 & 0xFFFF) << 16;                           // tensor_dim0[15:0]
      g1[2] = ((td0 >> 16) & 0xFFFF) | ((td1 & 0xFFFF) << 16);// td0 hi | td1 lo
      g1[3] = ((td1 >> 16) & 0xFFFF) | (BK << 16);            // td1 hi | tile_dim0
      g1[4] = BN;                                             // tile_dim1 rows
      g1[5] = Ktot;                                           // tensor_dim0_stride
      g1[6] = 0;
      g1[7] = 0;
      i32x4 z4; z4[0] = 0; z4[1] = 0; z4[2] = 0; z4[3] = 0;
#if defined(__clang_major__) && __clang_major__ >= 23
      i32x8 z8; for (int e = 0; e < 8; ++e) z8[e] = 0;
      __builtin_amdgcn_tensor_load_to_lds(g0, g1, z4, z4, z8, 0);
#else
      __builtin_amdgcn_tensor_load_to_lds(g0, g1, z4, z4, 0);
#endif
    }

    // ---- stage A tile (overlaps with the TDM transfer) ----
    {
      const int k = k0 + kk;
      const bool kval = (k < Ktot);
      const float* pb = nullptr;
      int dy = 0, dx = 0;
      if (kval) {
        int ci = k / 9;
        const int tap = k - ci * 9;
        dy = tap / 3 - 1;
        dx = tap - (tap / 3) * 3 - 1;
        const float* p; int cs, bs;
        if (ci < C0)       { p = p0; cs = cs0; bs = bs0; }
        else if (ci < C01) { ci -= C0;  p = p1; cs = cs1; bs = bs1; }
        else               { ci -= C01; p = p2; cs = cs2; bs = bs2; }
        pb = p + (size_t)bimg * bs + (size_t)ci * cs;
      }
      int pix = pix0 + rg;
      int y = pix / Wpx;
      int x = pix - y * Wpx;
      unsigned short* sa = &sA[rg * LDA + kk];
      for (int i = 0; i < 32; ++i) {
        const int yy = y + dy, xx = x + dx;
        float v = 0.f;
        if (kval && yy >= 0 && yy < Hn && xx >= 0 && xx < Wpx)
          v = pb[yy * Wpx + xx];
        sa[(i << 2) * LDA] = f2bf(v);
        x += 4;
        if (x >= Wpx) { x -= Wpx; ++y; }
      }
    }

    if (wave == 0) __builtin_amdgcn_s_wait_tensorcnt((short)0);
    __syncthreads();

    // ---- 2 x (K=32) WMMA sub-steps: 8 WMMAs per staged tile per wave ----
    {
      const int half = lane >> 4;
      const int l16  = lane & 15;
      union Frag { unsigned int u[8]; v16bf v; };
      for (int ks = 0; ks < 2; ++ks) {
        const int ko = ks * 32;
        Frag afr[2], bfr[2];
        for (int i = 0; i < 2; ++i) {
          const unsigned short* ap = &sA[(wm * 32 + i * 16 + l16) * LDA + ko];
          const int kb = half * 8;
          for (int j = 0; j < 4; ++j) {
            afr[i].u[j]     = *(const unsigned int*)&ap[kb + 2 * j];
            afr[i].u[4 + j] = *(const unsigned int*)&ap[16 + kb + 2 * j];
          }
        }
        for (int j2 = 0; j2 < 2; ++j2) {
          const unsigned short* bp =
              &sB[(wn * 32 + j2 * 16 + l16) * LDB + ko];
          for (int j = 0; j < 4; ++j) {
            bfr[j2].u[j]     = *(const unsigned int*)&bp[half * 8 + 2 * j];
            bfr[j2].u[4 + j] = *(const unsigned int*)&bp[16 + half * 8 + 2 * j];
          }
        }
        for (int i = 0; i < 2; ++i)
          for (int j = 0; j < 2; ++j)
            acc[i][j] = __builtin_amdgcn_wmma_f32_16x16x32_bf16(
                false, afr[i].v, false, bfr[j].v, (short)0, acc[i][j], false, false);
      }
    }
    __syncthreads();
  }

  // ---- epilogue: bias + activation, scatter to NCHW output ----
  const int half = lane >> 4;
  const int l16  = lane & 15;
  for (int i = 0; i < 2; ++i) {
    for (int j = 0; j < 2; ++j) {
      const int n = n0 + wn * 32 + j * 16 + l16;
      if (n >= N) continue;
      const float bv = bias ? bias[n] : 0.f;
      union { v8f v; float f[8]; } a2; a2.v = acc[i][j];
      float* op = out + (size_t)bimg * out_bs + (size_t)n * HWn;
      for (int v = 0; v < 8; ++v) {
        const int pix = pix0 + wm * 32 + i * 16 + half * 8 + v;
        float val = a2.f[v] + bv;
        if (act == 1)      val = tanhf(val);
        else if (act == 2) val = val > 0.f ? val : 0.01f * val;
        op[pix] = val;
      }
    }
  }
}

// ---------------------------------------------------------------------------
// Pointwise kernels
// ---------------------------------------------------------------------------
__global__ void cvt_bf16(const float* __restrict__ in, unsigned short* __restrict__ outp, int n) {
  int i = blockIdx.x * blockDim.x + threadIdx.x;
  if (i < n) outp[i] = f2bf(in[i]);
}

__global__ void zero_f(float* __restrict__ p, int n) {
  int i = blockIdx.x * blockDim.x + threadIdx.x;
  if (i < n) p[i] = 0.f;
}

// gates: (B, 4*HID, H, W) in i,f,o,g order; updates h,c in place; optional running sums
__global__ void lstm_point(const float* __restrict__ gates, float* __restrict__ h,
                           float* __restrict__ c, float* __restrict__ sh,
                           float* __restrict__ sc, int addsum)
{
  int idx = blockIdx.x * blockDim.x + threadIdx.x;
  if (idx >= Bn * HIDn * HWn) return;
  int b   = idx / (HIDn * HWn);
  int r   = idx - b * HIDn * HWn;
  int ch  = r / HWn;
  int pix = r - ch * HWn;
  size_t base = (size_t)b * 4 * HIDn * HWn + pix;
  float gi = gates[base + (size_t)ch * HWn];
  float gf = gates[base + (size_t)(HIDn + ch) * HWn];
  float go = gates[base + (size_t)(2 * HIDn + ch) * HWn];
  float gg = gates[base + (size_t)(3 * HIDn + ch) * HWn];
  float c2 = sigf(gf) * c[idx] + sigf(gi) * tanhf(gg);
  float hn = sigf(go) * tanhf(c2);
  c[idx] = c2; h[idx] = hn;
  if (addsum) { sh[idx] += hn; sc[idx] += c2; }
}

// softmax over D channels at each (b, pix), in place
__global__ void softmax_d(float* __restrict__ a) {
  int idx = blockIdx.x * blockDim.x + threadIdx.x;
  if (idx >= Bn * HWn) return;
  int b = idx / HWn, pix = idx - b * HWn;
  size_t base = (size_t)b * Dn * HWn + pix;
  float v[Dn];
  float mx = -1e30f;
  for (int d = 0; d < Dn; ++d) { v[d] = a[base + (size_t)d * HWn]; mx = fmaxf(mx, v[d]); }
  float s = 0.f;
  for (int d = 0; d < Dn; ++d) { v[d] = __expf(v[d] - mx); s += v[d]; }
  float inv = 1.f / s;
  for (int d = 0; d < Dn; ++d) a[base + (size_t)d * HWn] = v[d] * inv;
}

// x_att[b,t,d,pix] *= alpha[b,d,pix]
__global__ void scale_x(float* __restrict__ x_att, const float* __restrict__ alpha) {
  int idx = blockIdx.x * blockDim.x + threadIdx.x;
  if (idx >= Bn * TIN * Dn * HWn) return;
  int pix = idx % HWn;
  int t2  = idx / HWn;
  int d   = t2 % Dn;
  int t3  = t2 / Dn;
  int b   = t3 / TIN;
  x_att[idx] *= alpha[(size_t)(b * Dn + d) * HWn + pix];
}

// y0 = x[:, TIN-1, 0]
__global__ void y0_copy(const float* __restrict__ x, float* __restrict__ y) {
  int idx = blockIdx.x * blockDim.x + threadIdx.x;
  if (idx >= Bn * HWn) return;
  int b = idx / HWn, pix = idx - b * HWn;
  y[idx] = x[((size_t)(b * TIN + (TIN - 1)) * Dn + 0) * HWn + pix];
}

// d_out[b, t, 0, pix] = y[b, pix]
__global__ void y_out_copy(const float* __restrict__ y, float* __restrict__ outp, int t) {
  int idx = blockIdx.x * blockDim.x + threadIdx.x;
  if (idx >= Bn * HWn) return;
  int b = idx / HWn, pix = idx - b * HWn;
  outp[(size_t)(b * TOUT + t) * HWn + pix] = y[idx];
}

// ---------------------------------------------------------------------------
extern "C" void kernel_launch(void* const* d_in, const int* in_sizes, int n_in,
                              void* d_out, int out_size, void* d_ws, size_t ws_size,
                              hipStream_t stream)
{
  (void)in_sizes; (void)n_in; (void)out_size; (void)ws_size;

  const float* x      = (const float*)d_in[0];
  const float* h0_in  = (const float*)d_in[1];
  const float* c0_in  = (const float*)d_in[2];
  const float* h1_in  = (const float*)d_in[3];
  const float* c1_in  = (const float*)d_in[4];
  const float* attn_w1 = (const float*)d_in[5];
  const float* attn_b1 = (const float*)d_in[6];
  const float* attn_w2 = (const float*)d_in[7];
  const float* attn_b2 = (const float*)d_in[8];
  const float* enc_w0  = (const float*)d_in[9];
  const float* enc_b0  = (const float*)d_in[10];
  const float* enc_w1  = (const float*)d_in[11];
  const float* enc_b1  = (const float*)d_in[12];
  const float* dec_w0  = (const float*)d_in[13];
  const float* dec_b0  = (const float*)d_in[14];
  const float* dec_w1  = (const float*)d_in[15];
  const float* dec_b1  = (const float*)d_in[16];
  const float* out_w1  = (const float*)d_in[17];
  const float* out_w2  = (const float*)d_in[18];
  float* outp = (float*)d_out;

  const size_t NX = (size_t)Bn * TIN * Dn * HWn;      // attended input
  const size_t S  = (size_t)Bn * HIDn * HWn;          // one state tensor
  const size_t G  = (size_t)Bn * 4 * HIDn * HWn;      // gates

  float* f = (float*)d_ws;
  size_t off = 0;
  auto falloc = [&](size_t n) { float* p = f + off; off += n; return p; };
  float* x_att = falloc(NX);
  float* h0 = falloc(S); float* c0 = falloc(S);
  float* h1 = falloc(S); float* c1 = falloc(S);
  float* dh0 = falloc(S); float* dc0 = falloc(S);   // accumulate sum(h1_seq), sum(c1_seq)
  float* dh1 = falloc(S); float* dc1 = falloc(S);   // accumulate sum(h0_seq), sum(c0_seq)
  float* gates = falloc(G);
  float* z1    = falloc((size_t)Bn * 16 * HWn);
  float* aat   = falloc((size_t)Bn * Dn * HWn);
  float* ymid  = falloc((size_t)Bn * 16 * HWn);
  float* ybuf  = falloc((size_t)Bn * HWn);

  unsigned short* wb = (unsigned short*)(f + off);
  size_t woff = 0;
  auto walloc = [&](size_t n) { unsigned short* p = wb + woff; woff += n; return p; };
  const int nA1 = 16 * (TIN + 2 * HIDn) * 9;   // 20160
  const int nA2 = 1 * 16 * 9;                  // 144
  const int nE0 = 256 * (Dn + HIDn) * 9;       // 170496
  const int nE1 = 256 * (2 * HIDn) * 9;        // 294912
  const int nD0 = 256 * (1 + HIDn) * 9;        // 149760
  const int nD1 = 256 * (2 * HIDn) * 9;        // 294912
  const int nO1 = 16 * HIDn * 9;               // 9216
  const int nO2 = 1 * 16 * 9;                  // 144
  unsigned short* wbA1 = walloc(nA1);
  unsigned short* wbA2 = walloc(nA2);
  unsigned short* wbE0 = walloc(nE0);
  unsigned short* wbE1 = walloc(nE1);
  unsigned short* wbD0 = walloc(nD0);
  unsigned short* wbD1 = walloc(nD1);
  unsigned short* wbO1 = walloc(nO1);
  unsigned short* wbO2 = walloc(nO2);

  auto cvt = [&](const float* src, unsigned short* dst, int n) {
    cvt_bf16<<<(n + 255) / 256, 256, 0, stream>>>(src, dst, n);
  };
  cvt(attn_w1, wbA1, nA1); cvt(attn_w2, wbA2, nA2);
  cvt(enc_w0, wbE0, nE0);  cvt(enc_w1, wbE1, nE1);
  cvt(dec_w0, wbD0, nD0);  cvt(dec_w1, wbD1, nD1);
  cvt(out_w1, wbO1, nO1);  cvt(out_w2, wbO2, nO2);

  // init state
  hipMemcpyAsync(x_att, x, NX * sizeof(float), hipMemcpyDeviceToDevice, stream);
  hipMemcpyAsync(h0, h0_in, S * sizeof(float), hipMemcpyDeviceToDevice, stream);
  hipMemcpyAsync(c0, c0_in, S * sizeof(float), hipMemcpyDeviceToDevice, stream);
  hipMemcpyAsync(h1, h1_in, S * sizeof(float), hipMemcpyDeviceToDevice, stream);
  hipMemcpyAsync(c1, c1_in, S * sizeof(float), hipMemcpyDeviceToDevice, stream);
  const int zg = (int)((S + 255) / 256);
  zero_f<<<zg, 256, 0, stream>>>(dh0, (int)S);
  zero_f<<<zg, 256, 0, stream>>>(dc0, (int)S);
  zero_f<<<zg, 256, 0, stream>>>(dh1, (int)S);
  zero_f<<<zg, 256, 0, stream>>>(dc1, (int)S);

  auto conv = [&](const unsigned short* Wb, const float* bias, float* op, int out_bs,
                  int N, int Ktot, int act,
                  const float* p0, int C0, int cs0, int bs0,
                  const float* p1, int C1, int cs1, int bs1,
                  const float* p2, int C2, int cs2, int bs2) {
    dim3 g(Mtot / BM, (N + BN - 1) / BN);
    conv_wmma<<<g, 256, 0, stream>>>(Wb, bias, op, out_bs, N, Ktot, act,
                                     p0, C0, cs0, bs0, p1, C1, cs1, bs1, p2, C2, cs2, bs2);
  };

  const int lstm_g = (int)(S / 256);
  const int sm_g   = (Bn * HWn) / 256;
  const int sc_g   = (int)(NX / 256);

  // ===================== encoder =====================
  for (int t = 0; t < TIN; ++t) {
    // input attention: per-variable conv over concat(x[:,:,d], h0, c0)
    for (int d = 0; d < Dn; ++d) {
      conv(wbA1, attn_b1, z1, 16 * HWn, 16, (TIN + 2 * HIDn) * 9, /*tanh*/1,
           x_att + (size_t)d * HWn, TIN, Dn * HWn, TIN * Dn * HWn,
           h0, HIDn, HWn, HIDn * HWn,
           c0, HIDn, HWn, HIDn * HWn);
      conv(wbA2, attn_b2, aat + (size_t)d * HWn, Dn * HWn, 1, 16 * 9, 0,
           z1, 16, HWn, 16 * HWn,
           nullptr, 0, 0, 0, nullptr, 0, 0, 0);
    }
    softmax_d<<<sm_g, 256, 0, stream>>>(aat);
    scale_x<<<sc_g, 256, 0, stream>>>(x_att, aat);

    // encoder layer 0: conv over concat(x_att[:,t], h0)
    conv(wbE0, enc_b0, gates, 4 * HIDn * HWn, 4 * HIDn, (Dn + HIDn) * 9, 0,
         x_att + (size_t)t * Dn * HWn, Dn, HWn, TIN * Dn * HWn,
         h0, HIDn, HWn, HIDn * HWn,
         nullptr, 0, 0, 0);
    lstm_point<<<lstm_g, 256, 0, stream>>>(gates, h0, c0, dh1, dc1, 1); // sum h0_seq -> dec layer1

    // encoder layer 1: conv over concat(h0_new, h1)
    conv(wbE1, enc_b1, gates, 4 * HIDn * HWn, 4 * HIDn, (2 * HIDn) * 9, 0,
         h0, HIDn, HWn, HIDn * HWn,
         h1, HIDn, HWn, HIDn * HWn,
         nullptr, 0, 0, 0);
    lstm_point<<<lstm_g, 256, 0, stream>>>(gates, h1, c1, dh0, dc0, 1); // sum h1_seq -> dec layer0
  }

  // ===================== decoder =====================
  y0_copy<<<sm_g, 256, 0, stream>>>(x, ybuf);
  for (int t = 0; t < TOUT; ++t) {
    conv(wbD0, dec_b0, gates, 4 * HIDn * HWn, 4 * HIDn, (1 + HIDn) * 9, 0,
         ybuf, 1, HWn, HWn,
         dh0, HIDn, HWn, HIDn * HWn,
         nullptr, 0, 0, 0);
    lstm_point<<<lstm_g, 256, 0, stream>>>(gates, dh0, dc0, nullptr, nullptr, 0);

    conv(wbD1, dec_b1, gates, 4 * HIDn * HWn, 4 * HIDn, (2 * HIDn) * 9, 0,
         dh0, HIDn, HWn, HIDn * HWn,
         dh1, HIDn, HWn, HIDn * HWn,
         nullptr, 0, 0, 0);
    lstm_point<<<lstm_g, 256, 0, stream>>>(gates, dh1, dc1, nullptr, nullptr, 0);

    conv(wbO1, nullptr, ymid, 16 * HWn, 16, HIDn * 9, /*leaky*/2,
         dh1, HIDn, HWn, HIDn * HWn,
         nullptr, 0, 0, 0, nullptr, 0, 0, 0);
    conv(wbO2, nullptr, ybuf, HWn, 1, 16 * 9, /*leaky*/2,
         ymid, 16, HWn, 16 * HWn,
         nullptr, 0, 0, 0, nullptr, 0, 0, 0);
    y_out_copy<<<sm_g, 256, 0, stream>>>(ybuf, outp, t);
  }
}
